// Model_52518860096395
// MI455X (gfx1250) — matrix operand order repaired and checked by reading, not compiled, not run on hardware
//
#include <hip/hip_runtime.h>

typedef _Float16 half_t;
typedef __attribute__((ext_vector_type(16))) _Float16 v16h;
typedef __attribute__((ext_vector_type(8)))  float    v8f;
typedef unsigned __attribute__((ext_vector_type(4)))  u32x4;
typedef int      __attribute__((ext_vector_type(4)))  i32x4;
typedef int      __attribute__((ext_vector_type(8)))  i32x8;

union Frag16 {
  v16h  v;
  uint4 u[2];
  half_t h[16];
};

#define B_   8
#define H_   8
#define SQ_  2048
#define SK_  2048
#define DH_  64
#define DIN_ 32
#define DP_  512

// combined softmax scale: (1-dropout) / sqrt(head_dim) = 0.9 / 8
#define SM_SCALE 0.1125f

static __device__ __forceinline__ v8f wmma_f16(v16h a, v16h b, v8f c) {
  return __builtin_amdgcn_wmma_f32_16x16x32_f16(false, a, false, b, (short)0, c,
                                                false, false);
}

// ---------------------------------------------------------------------------
// TDM helper: issue one tensor-DMA tile load (global -> LDS).
// g0 packs D# group0: count=1, lds_addr, 57-bit global_addr, type=2.
// g1 (caller-supplied) packs D# group1: data_size / LDS padding / dims.
// Toolchain uses the 6-arg builtin: (g0, g1, g2, g3, g4, cpol).
// ---------------------------------------------------------------------------
static __device__ __forceinline__ unsigned lds_off(const void* p) {
  // generic pointer to __shared__: low 32 bits are the LDS byte offset
  return (unsigned)(unsigned long long)(uintptr_t)p;
}

static __device__ __forceinline__ void tdm_load(unsigned lds_addr,
                                                const void* gptr, i32x8 g1) {
  unsigned long long ga = (unsigned long long)(uintptr_t)gptr;
  u32x4 g0;
  g0[0] = 1u;                                   // count=1, no gather
  g0[1] = lds_addr;                             // LDS byte address
  g0[2] = (unsigned)ga;                         // global_addr[31:0]
  g0[3] = (unsigned)((ga >> 32) & 0x01FFFFFFu)  // global_addr[56:32]
          | (2u << 30);                         // type=2 ("image")
  i32x4 z4 = {0, 0, 0, 0};                      // groups 2/3 unused (2D tile)
  i32x8 z8 = {0, 0, 0, 0, 0, 0, 0, 0};
  __builtin_amdgcn_tensor_load_to_lds(g0, g1, z4, z4, z8, 0);
}

// ---------------------------------------------------------------------------
// Projection: out[n, m] = sum_c x[m,c] * W[n,c] + bias[n], f32 in -> f16 out.
// One wave per 16x16 output tile; K = 32 = D_in -> exactly one WMMA.
// vtrans==0: out layout [B,H,S,64]; vtrans==1: out layout [B,H,64,S]
// ---------------------------------------------------------------------------
__global__ __launch_bounds__(256) void proj_kernel(const float* __restrict__ x,
                                                   const float* __restrict__ W,
                                                   const float* __restrict__ bias,
                                                   half_t* __restrict__ out,
                                                   int vtrans) {
  const int tid  = threadIdx.x;
  const int lane = tid & 31;
  const int wid  = tid >> 5;
  const int t    = blockIdx.x * 8 + wid;   // tile id
  const int mt   = t >> 5;                 // 0..1023  (row tiles of 16)
  const int nt   = t & 31;                 // 0..31    (col tiles of 16)
  const int l15  = lane & 15;
  const int hi   = lane >> 4;
  const int lo8  = hi ? 8 : 0;
  const int lo16 = hi ? 16 : 0;

  // A fragment: 16x32 tile of x (f32 -> f16)
  const float* xr = x + (size_t)(mt * 16 + l15) * DIN_;
  Frag16 a;
  float4 f;
  f = *(const float4*)(xr + lo8);          a.h[0]=(half_t)f.x;  a.h[1]=(half_t)f.y;  a.h[2]=(half_t)f.z;  a.h[3]=(half_t)f.w;
  f = *(const float4*)(xr + lo8 + 4);      a.h[4]=(half_t)f.x;  a.h[5]=(half_t)f.y;  a.h[6]=(half_t)f.z;  a.h[7]=(half_t)f.w;
  f = *(const float4*)(xr + 16 + lo8);     a.h[8]=(half_t)f.x;  a.h[9]=(half_t)f.y;  a.h[10]=(half_t)f.z; a.h[11]=(half_t)f.w;
  f = *(const float4*)(xr + 16 + lo8 + 4); a.h[12]=(half_t)f.x; a.h[13]=(half_t)f.y; a.h[14]=(half_t)f.z; a.h[15]=(half_t)f.w;

  // B fragment: B(c,n) = W[n][c]; lane = column n, slots = c (lo8..+7 / 16+lo8..+7)
  const float* wr = W + (size_t)(nt * 16 + l15) * DIN_;
  Frag16 bf;
  f = *(const float4*)(wr + lo8);      bf.h[0]=(half_t)f.x;  bf.h[1]=(half_t)f.y;  bf.h[2]=(half_t)f.z;  bf.h[3]=(half_t)f.w;
  f = *(const float4*)(wr + lo8 + 4);  bf.h[4]=(half_t)f.x;  bf.h[5]=(half_t)f.y;  bf.h[6]=(half_t)f.z;  bf.h[7]=(half_t)f.w;
  f = *(const float4*)(wr + 16 + lo8);  bf.h[8]=(half_t)f.x;  bf.h[9]=(half_t)f.y;  bf.h[10]=(half_t)f.z; bf.h[11]=(half_t)f.w;
  f = *(const float4*)(wr + 16 + lo8 + 4); bf.h[12]=(half_t)f.x; bf.h[13]=(half_t)f.y; bf.h[14]=(half_t)f.z; bf.h[15]=(half_t)f.w;

  v8f acc = {0.f,0.f,0.f,0.f,0.f,0.f,0.f,0.f};
  acc = wmma_f16(a.v, bf.v, acc);

  const int   nn = nt * 16 + l15;          // 0..511
  const float bn = bias[nn];
  const int   hh = nn >> 6;
  const int   dd = nn & 63;

#pragma unroll
  for (int j = 0; j < 8; ++j) {
    const int m  = mt * 16 + j + (hi << 3);
    const int bb = m >> 11;
    const int ss = m & 2047;
    const float vjf = acc[j] + bn;
    size_t idx;
    if (vtrans) idx = ((size_t)(bb * H_ + hh) * DH_ + dd) * (size_t)SK_ + ss;
    else        idx = ((size_t)(bb * H_ + hh) * SQ_ + ss) * (size_t)DH_ + dd;
    out[idx] = (half_t)vjf;
  }
}

// ---------------------------------------------------------------------------
// Flash attention: one workgroup per (b,h, 128-query block); 8 waves, each
// wave owns 16 query rows. K/V staged in double-buffered LDS tiles of 32 keys
// via the Tensor Data Mover (one issuing wave, TENSORcnt + barrier).
// ---------------------------------------------------------------------------
#define KSTR 72   // halves per K-tile row (128B data + 16B TDM pad), 16B aligned
#define VSTR 40   // halves per Vt row (64B data + 16B TDM pad)
#define PSTR 40   // halves per P row

__global__ __launch_bounds__(256) void attn_kernel(const half_t* __restrict__ qh,
                                                   const half_t* __restrict__ kh,
                                                   const half_t* __restrict__ vt,
                                                   float* __restrict__ out) {
  __shared__ alignas(16) half_t lds_k[2][32 * KSTR];
  __shared__ alignas(16) half_t lds_v[2][64 * VSTR];
  __shared__ alignas(16) half_t lds_p[8][16 * PSTR];

  const int tid   = threadIdx.x;
  const int lane  = tid & 31;
  const int wid   = tid >> 5;
  const int bh    = blockIdx.x >> 4;       // b*8 + h
  const int qblk  = blockIdx.x & 15;
  const int qrow0 = qblk * 128 + wid * 16;
  const int l15   = lane & 15;
  const int hi    = lane >> 4;
  const int lo8   = hi ? 8 : 0;
  const int lo16  = hi ? 16 : 0;

  // ---- Q fragments (A-layout, 16 rows x 64 d), loaded once
  const half_t* qr = qh + ((size_t)bh * SQ_ + qrow0 + l15) * DH_;
  Frag16 aq0, aq1;
  aq0.u[0] = *(const uint4*)(qr + lo8);
  aq0.u[1] = *(const uint4*)(qr + 16 + lo8);
  aq1.u[0] = *(const uint4*)(qr + 32 + lo8);
  aq1.u[1] = *(const uint4*)(qr + 48 + lo8);

  const half_t* ksrc = kh + (size_t)bh * SK_ * DH_;       // [Sk][64]
  const half_t* vsrc = vt + (size_t)bh * DH_ * SK_;       // [64][Sk]

  // ---- TDM D# group1 descriptors (constant per kernel)
  // K tile: 32 rows x 64 halves; row = 32 DWORDs, pad 4 DWORDs -> KSTR=72
  //   data_size=2B (1<<16), pad_enable (1<<20), pad_interval=32dw (4<<22),
  //   pad_amount=4dw (3<<25)
  const i32x8 kdesc = { (1 << 16) | (1 << 20) | (4 << 22) | (3 << 25),
                        (DH_ << 16),        // tensor_dim0 = 64
                        (SK_ << 16),        // tensor_dim1 = 2048
                        (DH_ << 16),        // tile_dim0   = 64
                        32,                 // tile_dim1   = 32
                        DH_,                // tensor_dim0_stride = 64
                        0, 0 };
  // V tile: 64 rows x 32 halves from [64][2048]; row = 16 DWORDs, pad 4 DWORDs
  //   -> VSTR=40; pad_interval=16dw (3<<22)
  const i32x8 vdesc = { (1 << 16) | (1 << 20) | (3 << 22) | (3 << 25),
                        (SK_ << 16),        // tensor_dim0 = 2048
                        (DH_ << 16),        // tensor_dim1 = 64
                        (32 << 16),         // tile_dim0   = 32
                        DH_,                // tile_dim1   = 64
                        SK_,                // tensor_dim0_stride = 2048
                        0, 0 };

  // ---- preload tile 0 via TDM (wave 0 issues; TENSORcnt tracks completion)
  if (wid == 0) {
    tdm_load(lds_off(&lds_k[0][0]), ksrc, kdesc);
    tdm_load(lds_off(&lds_v[0][0]), vsrc, vdesc);
    __builtin_amdgcn_s_wait_tensorcnt((short)0);
  }
  __syncthreads();

  v8f o0 = {0.f,0.f,0.f,0.f,0.f,0.f,0.f,0.f};
  v8f o1 = o0, o2 = o0, o3 = o0;
  float mrow[8], lrow[8];
#pragma unroll
  for (int j = 0; j < 8; ++j) { mrow[j] = -3.0e38f; lrow[j] = 0.f; }

  const int NIT = SK_ / 32;                // 64 key tiles
  for (int it = 0; it < NIT; ++it) {
    const int  cur = it & 1;
    const bool has = (it + 1) < NIT;

    // ---- issue next tile's DMA before computing on the current one
    if (wid == 0 && has) {
      tdm_load(lds_off(&lds_k[cur ^ 1][0]),
               ksrc + (size_t)(it + 1) * 32 * DH_, kdesc);
      tdm_load(lds_off(&lds_v[cur ^ 1][0]),
               vsrc + (size_t)(it + 1) * 32, vdesc);
    }

    const half_t* kb = lds_k[cur];
    const half_t* vb = lds_v[cur];
    half_t*       pb = lds_p[wid];

    // ---- S = Q K^T over 32 keys (two 16-wide column tiles)
    v8f s0, s1;
    {
      const half_t* kr = kb + (size_t)l15 * KSTR;
      Frag16 b0, b1;
      b0.u[0] = *(const uint4*)(kr + lo8);
      b0.u[1] = *(const uint4*)(kr + 16 + lo8);
      b1.u[0] = *(const uint4*)(kr + 32 + lo8);
      b1.u[1] = *(const uint4*)(kr + 48 + lo8);
      v8f acc = {0.f,0.f,0.f,0.f,0.f,0.f,0.f,0.f};
      acc = wmma_f16(aq0.v, b0.v, acc);
      acc = wmma_f16(aq1.v, b1.v, acc);
      s0 = acc * SM_SCALE;
    }
    {
      const half_t* kr = kb + (size_t)(16 + l15) * KSTR;
      Frag16 b0, b1;
      b0.u[0] = *(const uint4*)(kr + lo8);
      b0.u[1] = *(const uint4*)(kr + 16 + lo8);
      b1.u[0] = *(const uint4*)(kr + 32 + lo8);
      b1.u[1] = *(const uint4*)(kr + 48 + lo8);
      v8f acc = {0.f,0.f,0.f,0.f,0.f,0.f,0.f,0.f};
      acc = wmma_f16(aq0.v, b0.v, acc);
      acc = wmma_f16(aq1.v, b1.v, acc);
      s1 = acc * SM_SCALE;
    }

    // ---- online softmax (row lives in one 16-lane half; C layout: row j+8*hi)
#pragma unroll
    for (int j = 0; j < 8; ++j) {
      float t = fmaxf(s0[j], s1[j]);
      t = fmaxf(t, __shfl_xor(t, 1, 32));
      t = fmaxf(t, __shfl_xor(t, 2, 32));
      t = fmaxf(t, __shfl_xor(t, 4, 32));
      t = fmaxf(t, __shfl_xor(t, 8, 32));
      const float mn = fmaxf(mrow[j], t);
      const float al = __expf(mrow[j] - mn);
      mrow[j] = mn;
      const float p0 = __expf(s0[j] - mn);
      const float p1 = __expf(s1[j] - mn);
      float r = p0 + p1;
      r += __shfl_xor(r, 1, 32);
      r += __shfl_xor(r, 2, 32);
      r += __shfl_xor(r, 4, 32);
      r += __shfl_xor(r, 8, 32);
      lrow[j] = lrow[j] * al + r;
      o0[j] *= al; o1[j] *= al; o2[j] *= al; o3[j] *= al;
      const int pr = (j + (hi << 3)) * PSTR;
      pb[pr + l15]      = (half_t)p0;
      pb[pr + 16 + l15] = (half_t)p1;
    }
    // wave-local C->A relayout through LDS; wait own DS stores (no WG barrier)
    asm volatile("s_wait_dscnt 0" ::: "memory");

    Frag16 pa;
    {
      const half_t* prd = pb + (size_t)l15 * PSTR;
      pa.u[0] = *(const uint4*)(prd + lo16);
      pa.u[1] = *(const uint4*)(prd + lo16 + 8);
    }

    // ---- O += P V   (B(kk,n) = V[kk][n] = Vt[n][kk], contiguous in LDS)
    {
      const half_t* vr = vb + (size_t)l15 * VSTR + lo16;
      Frag16 bv; bv.u[0] = *(const uint4*)(vr); bv.u[1] = *(const uint4*)(vr + 8);
      o0 = wmma_f16(pa.v, bv.v, o0);
    }
    {
      const half_t* vr = vb + (size_t)(16 + l15) * VSTR + lo16;
      Frag16 bv; bv.u[0] = *(const uint4*)(vr); bv.u[1] = *(const uint4*)(vr + 8);
      o1 = wmma_f16(pa.v, bv.v, o1);
    }
    {
      const half_t* vr = vb + (size_t)(32 + l15) * VSTR + lo16;
      Frag16 bv; bv.u[0] = *(const uint4*)(vr); bv.u[1] = *(const uint4*)(vr + 8);
      o2 = wmma_f16(pa.v, bv.v, o2);
    }
    {
      const half_t* vr = vb + (size_t)(48 + l15) * VSTR + lo16;
      Frag16 bv; bv.u[0] = *(const uint4*)(vr); bv.u[1] = *(const uint4*)(vr + 8);
      o3 = wmma_f16(pa.v, bv.v, o3);
    }

    // ---- DMA for next tile must be complete before all waves read it
    if (wid == 0 && has) {
      __builtin_amdgcn_s_wait_tensorcnt((short)0);
    }
    __syncthreads();
  }

  // ---- epilogue: divide by row sum, store f32 [B, Sq, H*64]
  const int bb = bh >> 3;
  const int hh = bh & 7;
#pragma unroll
  for (int j = 0; j < 8; ++j) {
    const int ss = qrow0 + j + (hi << 3);
    const float inv = 1.0f / lrow[j];
    float* op = out + ((size_t)bb * SQ_ + ss) * DP_ + hh * DH_ + l15;
    op[0]  = o0[j] * inv;
    op[16] = o1[j] * inv;
    op[32] = o2[j] * inv;
    op[48] = o3[j] * inv;
  }
}

// ---------------------------------------------------------------------------
extern "C" void kernel_launch(void* const* d_in, const int* in_sizes, int n_in,
                              void* d_out, int out_size, void* d_ws, size_t ws_size,
                              hipStream_t stream) {
  (void)in_sizes; (void)n_in; (void)out_size; (void)ws_size;
  const float* query = (const float*)d_in[0];
  const float* key   = (const float*)d_in[1];
  const float* value = (const float*)d_in[2];
  // d_in[3] = mask: only its shape feeds the reference (folded into SM_SCALE)
  const float* Wq = (const float*)d_in[4];
  const float* bq = (const float*)d_in[5];
  const float* Wk = (const float*)d_in[6];
  const float* bk = (const float*)d_in[7];
  const float* Wv = (const float*)d_in[8];
  const float* bv = (const float*)d_in[9];
  float* out = (float*)d_out;

  constexpr size_t PROJ_ELEMS = (size_t)B_ * H_ * SQ_ * DH_;  // 8388608
  half_t* qh  = (half_t*)d_ws;
  half_t* khp = qh + PROJ_ELEMS;
  half_t* vtp = khp + PROJ_ELEMS;

  // (B*S/16) * (512/16) tiles = 32768 waves -> 4096 blocks of 8 waves
  proj_kernel<<<dim3(4096), dim3(256), 0, stream>>>(query, Wq, bq, qh, 0);
  proj_kernel<<<dim3(4096), dim3(256), 0, stream>>>(key,   Wk, bk, khp, 0);
  proj_kernel<<<dim3(4096), dim3(256), 0, stream>>>(value, Wv, bv, vtp, 1);

  // B*H*(Sq/128) = 1024 workgroups
  attn_kernel<<<dim3(1024), dim3(256), 0, stream>>>(qh, khp, vtp, out);
}
